// ExpFloatLinear_38062000177375
// MI455X (gfx1250) — matrix operand order repaired and checked by reading, not compiled, not run on hardware
//
#include <hip/hip_runtime.h>
#include <hip/hip_bf16.h>

typedef __attribute__((ext_vector_type(16))) __bf16       v16bf;
typedef __attribute__((ext_vector_type(8)))  float        v8f;
typedef __attribute__((ext_vector_type(8)))  int          v8i;
typedef __attribute__((ext_vector_type(4)))  int          v4i;
typedef __attribute__((ext_vector_type(4)))  unsigned int v4u;

#ifndef __has_builtin
#define __has_builtin(x) 0
#endif

#if __has_builtin(__builtin_amdgcn_tensor_load_to_lds)
#define USE_TDM 1
#else
#define USE_TDM 0
#endif

#if __has_builtin(__builtin_amdgcn_global_load_async_to_lds_b128)
#define USE_ASYNC_LDS 1
#else
#define USE_ASYNC_LDS 0
#endif

#define AS1 __attribute__((address_space(1)))
#define AS3 __attribute__((address_space(3)))

// ---------------------------------------------------------------------------
// ExpFloat quant helpers: q(v) = 2^clamp(round(log2|v|), -128, 127), sign
// dropped; |v| <= 1e-38 -> 2^-128.  Powers of two are exact in f32 and bf16.
// ---------------------------------------------------------------------------
__device__ __forceinline__ float qd_f32(float v) {
    float a = __builtin_fabsf(v);
    float e;
    if (a > 1e-38f) {
        e = __builtin_rintf(__log2f(a));          // v_log_f32 + v_rndne_f32
        e = fminf(fmaxf(e, -128.0f), 127.0f);
    } else {
        e = -128.0f;
    }
    return __builtin_ldexpf(1.0f, (int)e);        // exact 2^e (denormal ok)
}

__device__ __forceinline__ unsigned short q_bf16_bits(float v) {
    // 2^e is exactly representable in bf16 down to 2^-133: truncate f32 bits.
    return (unsigned short)(__float_as_uint(qd_f32(v)) >> 16);
}

// ---------------------------------------------------------------------------
// Pass 1: quantize f32 tensor -> bf16 (power-of-two) tensor in workspace.
// ---------------------------------------------------------------------------
__global__ __launch_bounds__(256) void expq_kernel(
    const float* __restrict__ in, unsigned short* __restrict__ out) {
    const size_t idx = ((size_t)blockIdx.x * 256 + threadIdx.x) * 8;
    const float4* p = reinterpret_cast<const float4*>(in + idx);
    float4 v0 = p[0];
    float4 v1 = p[1];
    union { unsigned short s[8]; uint4 u; } r;
    r.s[0] = q_bf16_bits(v0.x); r.s[1] = q_bf16_bits(v0.y);
    r.s[2] = q_bf16_bits(v0.z); r.s[3] = q_bf16_bits(v0.w);
    r.s[4] = q_bf16_bits(v1.x); r.s[5] = q_bf16_bits(v1.y);
    r.s[6] = q_bf16_bits(v1.z); r.s[7] = q_bf16_bits(v1.w);
    *reinterpret_cast<uint4*>(out + idx) = r.u;
}

// ---------------------------------------------------------------------------
// Pass 2: NT GEMM  out[m,n] = qq( q(sum_k Aq[m,k]*Wq[n,k]) + q(bias[n]) )
// BM=BN=128, BK=64; 8 waves (wave32): wave grid 4(M) x 2(N), wave tile 32x64,
// 2x4 accumulators, 16 v_wmma_f32_16x16x32_bf16 per stage.  Double-buffered
// LDS tiles filled by the Tensor Data Mover (one TENSOR_LOAD_TO_LDS per tile,
// TENSORcnt) when available, else GLOBAL_LOAD_ASYNC_TO_LDS_B128 (ASYNCcnt).
// ---------------------------------------------------------------------------
#define BM 128
#define BN 128
#define BK 64

union FragBF { v16bf v; uint4 u[2]; };

#if USE_TDM
// One descriptor-driven DMA: 2D tile [rows x BK] bf16, row stride k_elems,
// global tile start -> LDS (rows stored contiguously, BK elems per row).
__device__ __forceinline__ void tdm_load_tile(const unsigned short* gbase,
                                              unsigned short* lbase,
                                              int rows, int k_elems) {
    const unsigned long long ga = (unsigned long long)gbase;
    const unsigned lds = (unsigned)(unsigned long long)(AS3 void*)lbase;
    union { v4u v; unsigned u[4]; } g0;
    g0.u[0] = 1u;                                     // count=1, user mode
    g0.u[1] = lds;                                    // lds_addr (bytes)
    g0.u[2] = (unsigned)ga;                           // global_addr[31:0]
    g0.u[3] = (unsigned)((ga >> 32) & 0x1FFFFFFu)     // global_addr[56:32]
              | (2u << 30);                           // type=2 ("image")
    union { v8i v; unsigned u[8]; } g1;
    g1.u[0] = 1u << 16;                               // data_size=1 (2 bytes)
    g1.u[1] = ((unsigned)k_elems & 0xFFFFu) << 16;    // tensor_dim0[15:0]
    g1.u[2] = ((unsigned)k_elems >> 16) & 0xFFFFu;    // tensor_dim0[31:16]
    g1.u[3] = ((unsigned)BK << 16) | 0x10u;           // tile_dim0=BK, dim1=2^20
    g1.u[4] = (unsigned)rows;                         // tile_dim1, tile_dim2=0
    g1.u[5] = (unsigned)k_elems;                      // dim0_stride[31:0]
    g1.u[6] = 0;                                      // stride hi, dim1_stride lo
    g1.u[7] = 0;
    v4i z4 = (v4i){0, 0, 0, 0};
    v8i z8 = (v8i){0, 0, 0, 0, 0, 0, 0, 0};
    __builtin_amdgcn_tensor_load_to_lds(g0.v, g1.v, z4, z4, z8, 0);
}
#endif

__device__ __forceinline__ void lds_async_b128(const unsigned short* g,
                                               unsigned short* l) {
#if USE_ASYNC_LDS
    __builtin_amdgcn_global_load_async_to_lds_b128(
        (AS1 v4i*)g, (AS3 v4i*)l, 0, 0);
#endif
}

__device__ __forceinline__ void wait_tensor0() {
#if __has_builtin(__builtin_amdgcn_s_wait_tensorcnt)
    __builtin_amdgcn_s_wait_tensorcnt(0);
#else
    asm volatile("s_wait_tensorcnt 0x0" ::: "memory");
#endif
}

__global__ __launch_bounds__(256) void expq_gemm_kernel(
    const unsigned short* __restrict__ Aq,   // [M][K] bf16 bits (quantized x)
    const unsigned short* __restrict__ Wq,   // [N][K] bf16 bits (quantized W)
    const float* __restrict__ bias,          // [N]
    float* __restrict__ out,                 // [M][N]
    int M, int N, int K) {
    __shared__ __align__(16) unsigned short As[2][BM][BK];   // 32 KB
    __shared__ __align__(16) unsigned short Bs[2][BN][BK];   // 32 KB

    const int tid  = threadIdx.x;
    const int lane = tid & 31;
    const int wave = tid >> 5;
    const int wm   = wave & 3;    // wave M index: offset wm*32
    const int wn   = wave >> 2;   // wave N index: offset wn*64
    const int lr   = lane & 15;
    const int lh   = lane >> 4;   // lane half (0/1)

    const int bm = blockIdx.y * BM;
    const int bn = blockIdx.x * BN;

    // per-thread mapping used by the async/sync fallback + L2 prefetch
    const int ldr = tid >> 1;            // row 0..127
    const int ldc = (tid & 1) * 32;      // element offset 0 / 32
    const unsigned short* Ag = Aq + (size_t)(bm + ldr) * K + ldc;
    const unsigned short* Bg = Wq + (size_t)(bn + ldr) * K + ldc;

#if USE_TDM
    const unsigned short* Atile = Aq + (size_t)bm * K;   // tile row 0
    const unsigned short* Btile = Wq + (size_t)bn * K;
#endif

    auto stage_load = [&](int buf, int k0) {
#if USE_TDM
        if (wave == 0) {
            tdm_load_tile(Atile + k0, &As[buf][0][0], BM, K);
        } else if (wave == 1) {
            tdm_load_tile(Btile + k0, &Bs[buf][0][0], BN, K);
        }
#elif USE_ASYNC_LDS
        const unsigned short* ga = Ag + k0;
        const unsigned short* gb = Bg + k0;
        unsigned short* la = &As[buf][ldr][ldc];
        unsigned short* lb = &Bs[buf][ldr][ldc];
        #pragma unroll
        for (int c = 0; c < 4; ++c) lds_async_b128(ga + c * 8, la + c * 8);
        #pragma unroll
        for (int c = 0; c < 4; ++c) lds_async_b128(gb + c * 8, lb + c * 8);
#else
        const unsigned short* ga = Ag + k0;
        const unsigned short* gb = Bg + k0;
        unsigned short* la = &As[buf][ldr][ldc];
        unsigned short* lb = &Bs[buf][ldr][ldc];
        uint4 ar[4], br[4];
        #pragma unroll
        for (int c = 0; c < 4; ++c) ar[c] = *reinterpret_cast<const uint4*>(ga + c * 8);
        #pragma unroll
        for (int c = 0; c < 4; ++c) br[c] = *reinterpret_cast<const uint4*>(gb + c * 8);
        #pragma unroll
        for (int c = 0; c < 4; ++c) *reinterpret_cast<uint4*>(la + c * 8) = ar[c];
        #pragma unroll
        for (int c = 0; c < 4; ++c) *reinterpret_cast<uint4*>(lb + c * 8) = br[c];
#endif
    };

    auto stage_wait = [&]() {
#if USE_TDM
        if (wave < 2) wait_tensor0();
#elif USE_ASYNC_LDS
        asm volatile("s_wait_asynccnt 0x0" ::: "memory");
#endif
    };

    v8f acc[2][4];
    #pragma unroll
    for (int i = 0; i < 2; ++i)
        #pragma unroll
        for (int j = 0; j < 4; ++j)
            #pragma unroll
            for (int e = 0; e < 8; ++e)
                acc[i][j][e] = 0.0f;

    stage_load(0, 0);
    stage_wait();
    __syncthreads();

    const int nk = K / BK;
    for (int kt = 0; kt < nk; ++kt) {
        const int cur = kt & 1;
        if (kt + 1 < nk) stage_load(cur ^ 1, (kt + 1) * BK);  // DMA next stage
        if (kt + 2 < nk) {                                    // L2 prefetch k+2
            __builtin_prefetch(Ag + (kt + 2) * BK, 0, 3);
            __builtin_prefetch(Bg + (kt + 2) * BK, 0, 3);
        }

        // Compute current stage: two 32-wide K steps.
        #pragma unroll
        for (int kk = 0; kk < BK; kk += 32) {
            FragBF af[2];
            #pragma unroll
            for (int i = 0; i < 2; ++i) {
                const int row = wm * 32 + i * 16 + lr;
                af[i].u[0] = *reinterpret_cast<const uint4*>(&As[cur][row][kk + lh * 8]);
                af[i].u[1] = *reinterpret_cast<const uint4*>(&As[cur][row][kk + 16 + lh * 8]);
            }
            FragBF bfr[4];
            #pragma unroll
            for (int j = 0; j < 4; ++j) {
                const int row = wn * 64 + j * 16 + lr;
                bfr[j].u[0] = *reinterpret_cast<const uint4*>(&Bs[cur][row][kk + lh * 8]);
                bfr[j].u[1] = *reinterpret_cast<const uint4*>(&Bs[cur][row][kk + 16 + lh * 8]);
            }
            #pragma unroll
            for (int i = 0; i < 2; ++i)
                #pragma unroll
                for (int j = 0; j < 4; ++j)
                    acc[i][j] = __builtin_amdgcn_wmma_f32_16x16x32_bf16(
                        false, af[i].v, false, bfr[j].v,
                        (short)0, acc[i][j], false, false);
        }

        stage_wait();         // my DMA writes to buf cur^1 landed in LDS
        __syncthreads();      // everyone done reading cur / writing cur^1
    }

    // Epilogue: out = q( q(acc) + q(bias) )
    const int mbase = bm + wm * 32 + lh * 8;
    #pragma unroll
    for (int j = 0; j < 4; ++j) {
        const int col = bn + wn * 64 + j * 16 + lr;
        const float bq = qd_f32(bias[col]);
        #pragma unroll
        for (int i = 0; i < 2; ++i) {
            float* op = out + (size_t)(mbase + i * 16) * N + col;
            #pragma unroll
            for (int v = 0; v < 8; ++v) {
                op[(size_t)v * N] = qd_f32(qd_f32(acc[i][j][v]) + bq);
            }
        }
    }
}

// ---------------------------------------------------------------------------
// Launch
// ---------------------------------------------------------------------------
extern "C" void kernel_launch(void* const* d_in, const int* in_sizes, int n_in,
                              void* d_out, int out_size, void* d_ws, size_t ws_size,
                              hipStream_t stream) {
    const float* x    = (const float*)d_in[0];   // [8192, 4096]
    const float* w    = (const float*)d_in[1];   // [4096, 4096]
    const float* bias = (const float*)d_in[2];   // [4096]
    float* out = (float*)d_out;

    const int M = 8192, N = 4096, K = 4096;

    unsigned short* xq = (unsigned short*)d_ws;            // 64 MB bf16
    unsigned short* wq = xq + (size_t)M * K;               // 32 MB bf16

    expq_kernel<<<(M * K) / (256 * 8), 256, 0, stream>>>(x, xq);
    expq_kernel<<<(N * K) / (256 * 8), 256, 0, stream>>>(w, wq);

    dim3 grid(N / BN, M / BM);
    expq_gemm_kernel<<<grid, 256, 0, stream>>>(xq, wq, bias, out, M, N, K);
}
